// MultiScaleFeatureEngine_77481210020608
// MI455X (gfx1250) — compile-verified, matrix operand order checked by reading, not run on hardware
//
#include <hip/hip_runtime.h>
#include <math.h>

// ---------------------------------------------------------------------------
// MultiScaleFeatureEngine fused kernel for gfx1250 (MI455X, wave32, WMMA).
//   B=65536, IN_DIM=256, D=32.  Memory-bound (~24 FLOP/B vs 23.3 TB/s HBM),
//   so: one pass over inputs, all GEMMs via v_wmma_f32_16x16x32_bf16,
//   LN/attention fused in-register using half-wave (16-lane) reductions.
//   This revision: ext-vector f32x4 for nontemporal b128 loads (HIP float4
//   is a struct and is rejected by __builtin_nontemporal_load).
// ---------------------------------------------------------------------------

typedef __attribute__((ext_vector_type(16))) __bf16   v16bf;
typedef __attribute__((ext_vector_type(2)))  __bf16   bf16x2;
typedef __attribute__((ext_vector_type(8)))  float    v8f;
typedef __attribute__((ext_vector_type(4)))  float    f32x4;
typedef __attribute__((ext_vector_type(8)))  unsigned v8u;
typedef __attribute__((ext_vector_type(4)))  unsigned u32x4;

#define IN_DIM 256
#define EMB    32
#define LN_EPS 1e-5f

// two f32 -> packed bf16 dword (matches v_cvt_pk_bf16_f32 when available)
__device__ __forceinline__ unsigned pk2(float x, float y) {
  bf16x2 p = { (__bf16)x, (__bf16)y };
  return __builtin_bit_cast(unsigned, p);
}

// sum across the 16-lane half-wave that holds one matrix row in C/D layout
__device__ __forceinline__ float red16(float x) {
#pragma unroll
  for (int m = 1; m < 16; m <<= 1) x += __shfl_xor(x, m, 16);
  return x;
}

// Assemble a 16-bit A/B fragment from 16 f32 covering this lane's K values:
//   q0 = K khalf+0..3, q1 = khalf+4..7, q2 = khalf+16..19, q3 = khalf+20..23
// VGPR v holds K pairs (0,1)(2,3)(4,5)(6,7)(16,17)(18,19)(20,21)(22,23)+khalf.
__device__ __forceinline__ v16bf frag16(f32x4 q0, f32x4 q1, f32x4 q2, f32x4 q3) {
  v8u w;
  w[0] = pk2(q0.x, q0.y); w[1] = pk2(q0.z, q0.w);
  w[2] = pk2(q1.x, q1.y); w[3] = pk2(q1.z, q1.w);
  w[4] = pk2(q2.x, q2.y); w[5] = pk2(q2.z, q2.w);
  w[6] = pk2(q3.x, q3.y); w[7] = pk2(q3.z, q3.w);
  return __builtin_bit_cast(v16bf, w);
}

// A/B fragment from f32 memory: base = &M[row_or_col][k0] for this lane
// (k0 = chunk base + khalf); two contiguous 32B runs -> 4x global_load_b128.
__device__ __forceinline__ v16bf load_frag_nt(const float* __restrict__ p) {
  f32x4 q0 = __builtin_nontemporal_load(reinterpret_cast<const f32x4*>(p + 0));
  f32x4 q1 = __builtin_nontemporal_load(reinterpret_cast<const f32x4*>(p + 4));
  f32x4 q2 = __builtin_nontemporal_load(reinterpret_cast<const f32x4*>(p + 16));
  f32x4 q3 = __builtin_nontemporal_load(reinterpret_cast<const f32x4*>(p + 20));
  return frag16(q0, q1, q2, q3);
}
__device__ __forceinline__ v16bf load_frag_rt(const float* __restrict__ p) {
  f32x4 q0 = *reinterpret_cast<const f32x4*>(p + 0);
  f32x4 q1 = *reinterpret_cast<const f32x4*>(p + 4);
  f32x4 q2 = *reinterpret_cast<const f32x4*>(p + 16);
  f32x4 q3 = *reinterpret_cast<const f32x4*>(p + 20);
  return frag16(q0, q1, q2, q3);
}

// A fragment from a wave-private LDS tile [16][32] bf16: 2x ds_load_b128
__device__ __forceinline__ v16bf load_a_lds(const __bf16* t, int lane) {
  const int row   = lane & 15;
  const int khalf = (lane >> 4) * 8;
  const __bf16* p = t + row * EMB + khalf;
  u32x4 a = *reinterpret_cast<const u32x4*>(p);
  u32x4 b = *reinterpret_cast<const u32x4*>(p + 16);
  v8u w;
  w[0] = a.x; w[1] = a.y; w[2] = a.z; w[3] = a.w;
  w[4] = b.x; w[5] = b.y; w[6] = b.z; w[7] = b.w;
  return __builtin_bit_cast(v16bf, w);
}

__device__ __forceinline__ v8f wmma_bf16(v16bf a, v16bf b, v8f c) {
  // D = A(16x32) * B(32x16) + C, f32 accumulate
  return __builtin_amdgcn_wmma_f32_16x16x32_bf16(false, a, false, b, (short)0, c, false, false);
}

__global__ void __launch_bounds__(128)
msfe_kernel(const float* __restrict__ f1, const float* __restrict__ f4,
            const float* __restrict__ fD,
            const float* __restrict__ Wp,  const float* __restrict__ bp,
            const float* __restrict__ ln_g, const float* __restrict__ ln_b,
            const float* __restrict__ Wq,  const float* __restrict__ bq,
            const float* __restrict__ Wk,  const float* __restrict__ bk,
            const float* __restrict__ Wv,  const float* __restrict__ bv,
            const float* __restrict__ Wo1, const float* __restrict__ bo1,
            const float* __restrict__ Wo2, const float* __restrict__ bo2,
            float* __restrict__ out)
{
  __shared__ __align__(16) __bf16 lds[4][3][16 * EMB];   // wave-private tiles

  const int lane = threadIdx.x & 31;
  const int wave = threadIdx.x >> 5;
  const int row0 = blockIdx.x * 64 + wave * 16;          // 16 batch rows per wave

  const int col   = lane & 15;      // N index within a 16-wide tile (C layout)
  const int half  = lane >> 4;      // C layout: row = r + 8*half
  const int khalf = half * 8;       // A/B layout K split

  // per-lane base pointers (row/col + khalf already folded in)
  const float* a0 = f1 + (size_t)(row0 + col) * IN_DIM + khalf;
  const float* a1 = f4 + (size_t)(row0 + col) * IN_DIM + khalf;
  const float* a2 = fD + (size_t)(row0 + col) * IN_DIM + khalf;
  const float* wp0 = Wp + (size_t)col        * IN_DIM + khalf;   // nt = 0
  const float* wp1 = Wp + (size_t)(16 + col) * IN_DIM + khalf;   // nt = 1

  // ---------------- Stage 1: projection GEMM [16x256]x[256x32] ----------------
  v8f accX[3][2];
#pragma unroll
  for (int s = 0; s < 3; ++s)
#pragma unroll
    for (int nt = 0; nt < 2; ++nt)
      accX[s][nt] = {};

#pragma unroll
  for (int kc = 0; kc < 8; ++kc) {
    v16bf b0 = load_frag_rt(wp0 + kc * 32);
    v16bf b1 = load_frag_rt(wp1 + kc * 32);
    {
      v16bf a = load_frag_nt(a0 + kc * 32);
      accX[0][0] = wmma_bf16(a, b0, accX[0][0]);
      accX[0][1] = wmma_bf16(a, b1, accX[0][1]);
    }
    {
      v16bf a = load_frag_nt(a1 + kc * 32);
      accX[1][0] = wmma_bf16(a, b0, accX[1][0]);
      accX[1][1] = wmma_bf16(a, b1, accX[1][1]);
    }
    {
      v16bf a = load_frag_nt(a2 + kc * 32);
      accX[2][0] = wmma_bf16(a, b0, accX[2][0]);
      accX[2][1] = wmma_bf16(a, b1, accX[2][1]);
    }
  }

  // per-lane params for the two column tiles this lane owns
  const float bp0 = bp[col],   bp1 = bp[16 + col];
  const float g0  = ln_g[col], g1  = ln_g[16 + col];
  const float be0 = ln_b[col], be1 = ln_b[16 + col];

  // ---------------- bias + ReLU + LayerNorm (in-register), X -> LDS -----------
#pragma unroll
  for (int s = 0; s < 3; ++s) {
    __bf16* xt = lds[wave][s];
#pragma unroll
    for (int r = 0; r < 8; ++r) {
      float h0 = fmaxf(accX[s][0][r] + bp0, 0.0f);
      float h1 = fmaxf(accX[s][1][r] + bp1, 0.0f);
      float mean = red16(h0 + h1) * (1.0f / 32.0f);
      float var  = red16(h0 * h0 + h1 * h1) * (1.0f / 32.0f) - mean * mean;
      float rstd = rsqrtf(var + LN_EPS);
      float x0 = (h0 - mean) * rstd * g0 + be0;
      float x1 = (h1 - mean) * rstd * g1 + be1;
      int row = r + half * 8;
      xt[row * EMB + col]      = (__bf16)x0;
      xt[row * EMB + 16 + col] = (__bf16)x1;
    }
  }

  // ---------------- Stage 2: Q/K/V = X @ W^T + b (WMMA, K=32) -----------------
  const float* wq0 = Wq + (size_t)col * EMB + khalf;  const float* wq1 = wq0 + 16 * EMB;
  const float* wk0 = Wk + (size_t)col * EMB + khalf;  const float* wk1 = wk0 + 16 * EMB;
  const float* wv0 = Wv + (size_t)col * EMB + khalf;  const float* wv1 = wv0 + 16 * EMB;
  v16bf bWq[2] = { load_frag_rt(wq0), load_frag_rt(wq1) };
  v16bf bWk[2] = { load_frag_rt(wk0), load_frag_rt(wk1) };
  v16bf bWv[2] = { load_frag_rt(wv0), load_frag_rt(wv1) };
  const float bqv[2] = { bq[col], bq[16 + col] };
  const float bkv[2] = { bk[col], bk[16 + col] };
  const float bvv[2] = { bv[col], bv[16 + col] };

  v8f Q[3][2], K[3][2], V[3][2];
#pragma unroll
  for (int s = 0; s < 3; ++s) {
    v16bf a = load_a_lds(lds[wave][s], lane);
#pragma unroll
    for (int nt = 0; nt < 2; ++nt) {
      v8f z = {};
      Q[s][nt] = wmma_bf16(a, bWq[nt], z);
      K[s][nt] = wmma_bf16(a, bWk[nt], z);
      V[s][nt] = wmma_bf16(a, bWv[nt], z);
#pragma unroll
      for (int r = 0; r < 8; ++r) {
        Q[s][nt][r] += bqv[nt];
        K[s][nt][r] += bkv[nt];
        V[s][nt][r] += bvv[nt];
      }
    }
  }

  // ---------------- Stage 3: 3x3 attention + mean pool (in-register) ----------
  v8f pooled[2];
  pooled[0] = {}; pooled[1] = {};
  const float scale = 0.17677669529663689f;  // 1/sqrt(32)
#pragma unroll
  for (int s = 0; s < 3; ++s) {
#pragma unroll
    for (int r = 0; r < 8; ++r) {
      float sc0 = red16(Q[s][0][r] * K[0][0][r] + Q[s][1][r] * K[0][1][r]) * scale;
      float sc1 = red16(Q[s][0][r] * K[1][0][r] + Q[s][1][r] * K[1][1][r]) * scale;
      float sc2 = red16(Q[s][0][r] * K[2][0][r] + Q[s][1][r] * K[2][1][r]) * scale;
      float m  = fmaxf(sc0, fmaxf(sc1, sc2));
      float e0 = __expf(sc0 - m), e1 = __expf(sc1 - m), e2 = __expf(sc2 - m);
      float inv = 1.0f / (e0 + e1 + e2);
      float w0 = e0 * inv, w1 = e1 * inv, w2 = e2 * inv;
      pooled[0][r] += (w0 * V[0][0][r] + w1 * V[1][0][r] + w2 * V[2][0][r]) * (1.0f / 3.0f);
      pooled[1][r] += (w0 * V[0][1][r] + w1 * V[1][1][r] + w2 * V[2][1][r]) * (1.0f / 3.0f);
    }
  }

  // ---------------- Stage 4: out = relu(pooled@Wo1^T+bo1)@Wo2^T+bo2+pooled ----
  const float* wo10 = Wo1 + (size_t)col * EMB + khalf;  const float* wo11 = wo10 + 16 * EMB;
  const float* wo20 = Wo2 + (size_t)col * EMB + khalf;  const float* wo21 = wo20 + 16 * EMB;
  v16bf bWo1[2] = { load_frag_rt(wo10), load_frag_rt(wo11) };
  v16bf bWo2[2] = { load_frag_rt(wo20), load_frag_rt(wo21) };
  const float bo1v[2] = { bo1[col], bo1[16 + col] };
  const float bo2v[2] = { bo2[col], bo2[16 + col] };

  // pooled (C layout) -> LDS tile 0 -> A layout
  __bf16* pt = lds[wave][0];
#pragma unroll
  for (int r = 0; r < 8; ++r) {
    int row = r + half * 8;
    pt[row * EMB + col]      = (__bf16)pooled[0][r];
    pt[row * EMB + 16 + col] = (__bf16)pooled[1][r];
  }
  v16bf ap = load_a_lds(pt, lane);
  v8f h[2];
  { v8f z = {}; h[0] = wmma_bf16(ap, bWo1[0], z); h[1] = wmma_bf16(ap, bWo1[1], z); }

  // relu(h) -> LDS tile 1 -> A layout
  __bf16* ht = lds[wave][1];
#pragma unroll
  for (int nt = 0; nt < 2; ++nt)
#pragma unroll
    for (int r = 0; r < 8; ++r) {
      float hv = fmaxf(h[nt][r] + bo1v[nt], 0.0f);
      int row = r + half * 8;
      ht[row * EMB + nt * 16 + col] = (__bf16)hv;
    }
  v16bf ah = load_a_lds(ht, lane);
  v8f o[2];
  { v8f z = {}; o[0] = wmma_bf16(ah, bWo2[0], z); o[1] = wmma_bf16(ah, bWo2[1], z); }

  // residual + bias, nontemporal store (rows coalesce across each 16-lane half)
#pragma unroll
  for (int nt = 0; nt < 2; ++nt)
#pragma unroll
    for (int r = 0; r < 8; ++r) {
      int row = r + half * 8;
      float v = o[nt][r] + bo2v[nt] + pooled[nt][r];
      __builtin_nontemporal_store(v, out + (size_t)(row0 + row) * EMB + nt * 16 + col);
    }
}

extern "C" void kernel_launch(void* const* d_in, const int* in_sizes, int n_in,
                              void* d_out, int out_size, void* d_ws, size_t ws_size,
                              hipStream_t stream) {
  const float* f1   = (const float*)d_in[0];
  const float* f4   = (const float*)d_in[1];
  const float* fD   = (const float*)d_in[2];
  const float* Wp   = (const float*)d_in[3];
  const float* bp   = (const float*)d_in[4];
  const float* ln_g = (const float*)d_in[5];
  const float* ln_b = (const float*)d_in[6];
  const float* Wq   = (const float*)d_in[7];
  const float* bq   = (const float*)d_in[8];
  const float* Wk   = (const float*)d_in[9];
  const float* bk   = (const float*)d_in[10];
  const float* Wv   = (const float*)d_in[11];
  const float* bv   = (const float*)d_in[12];
  const float* Wo1  = (const float*)d_in[13];
  const float* bo1  = (const float*)d_in[14];
  const float* Wo2  = (const float*)d_in[15];
  const float* bo2  = (const float*)d_in[16];

  const int B = in_sizes[0] / IN_DIM;      // 65536
  dim3 grid(B / 64), block(128);           // 4 waves/block, 16 rows/wave
  hipLaunchKernelGGL(msfe_kernel, grid, block, 0, stream,
                     f1, f4, fD, Wp, bp, ln_g, ln_b,
                     Wq, bq, Wk, bk, Wv, bv, Wo1, bo1, Wo2, bo2,
                     (float*)d_out);
}
